// TidalGeometricAttention_18339510354283
// MI455X (gfx1250) — matrix operand order repaired and checked by reading, not compiled, not run on hardware
//
#include <hip/hip_runtime.h>
#include <hip/hip_bf16.h>
#include <cstdint>

// Problem geometry (from reference): x[16384,512], positions[4096,512],
// values[4096,512], out[16384,512].
#define D_DIM 512
#define MB 128          // rows per workgroup (8 waves x 16 rows)
#define NB 128          // N-tile width
#define KB 64           // K staging chunk for GEMM1
#define DB 64           // D staging chunk for GEMM2
#define MIN_SCALE 0.05f

typedef __attribute__((ext_vector_type(16))) __bf16 bf16x16;
typedef __attribute__((ext_vector_type(8)))  float  f32x8;
typedef __attribute__((ext_vector_type(4)))  unsigned int u32x4;
typedef __attribute__((ext_vector_type(8)))  int  i32x8;
typedef __attribute__((ext_vector_type(4)))  int  i32x4;

// ---- Tensor Data Mover: 2D bf16 tile load, global -> LDS ------------------
// D# per CDNA5 ISA (§8.3 group0, §8.4 group1):
//  g0: [1:0]count=1 | [63:32]lds_addr | [120:64]global_addr | [127:126]type=2
//  g1: [17:16]data_size=1(2B) | [79:48]tensor_dim0 | [111:80]tensor_dim1 |
//      [127:112]tile_dim0 | [143:128]tile_dim1 | [207:160]tensor_dim0_stride
__device__ __forceinline__ void tdm_load_2d_bf16(const __bf16* gsrc,
                                                 unsigned lds_off,
                                                 unsigned tensor_w,
                                                 unsigned tensor_h,
                                                 unsigned tile_w,
                                                 unsigned tile_h) {
  unsigned long long ga = (unsigned long long)(uintptr_t)gsrc;
  u32x4 g0 = { 1u, lds_off, (unsigned)ga,
               ((unsigned)(ga >> 32) & 0x01FFFFFFu) | 0x80000000u };
  i32x8 g1 = { (int)(1u << 16),                                   // data_size=2B
               (int)((tensor_w & 0xFFFFu) << 16),                 // td0 lo16
               (int)((tensor_w >> 16) | ((tensor_h & 0xFFFFu) << 16)),
               (int)((tensor_h >> 16) | ((tile_w & 0xFFFFu) << 16)),
               (int)(tile_h & 0xFFFFu),                           // tile_dim1
               (int)tensor_w,                                     // dim0 stride lo32
               0, 0 };
  i32x4 g2 = { 1, 0, 0, 0 };   // tensor_dim2=1 (unused: tile_dim2=0)
  i32x4 g3 = { 0, 0, 0, 0 };
#if defined(__clang_major__) && (__clang_major__ >= 23)
  i32x8 g4 = { 0, 0, 0, 0, 0, 0, 0, 0 };
  __builtin_amdgcn_tensor_load_to_lds(g0, g1, g2, g3, g4, 0);
#else
  __builtin_amdgcn_tensor_load_to_lds(g0, g1, g2, g3, 0);
#endif
}

// ---- WMMA fragment loads from LDS (CDNA5 16-bit layouts, wave32) ----------
__device__ __forceinline__ bf16x16 lds_fragA(const __bf16* row, int k0, int h) {
  union { uint4 u[2]; bf16x16 v; } r;
  r.u[0] = *(const uint4*)(row + k0 + h * 8);
  r.u[1] = *(const uint4*)(row + k0 + 16 + h * 8);
  return r.v;
}
__device__ __forceinline__ bf16x16 lds_fragB(const __bf16* col, int k0, int h) {
  union { uint4 u[2]; bf16x16 v; } r;
  const __bf16* p = col + k0 + h * 8;
  r.u[0] = *(const uint4*)(p);
  r.u[1] = *(const uint4*)(p + 16);
  return r.v;
}

// ---- Preludes -------------------------------------------------------------
__global__ void tidal_rownorm(const float* __restrict__ src,
                              float* __restrict__ dst, int rows) {
  int wid  = (blockIdx.x * blockDim.x + threadIdx.x) >> 5;
  int lane = threadIdx.x & 31;
  if (wid >= rows) return;
  const float4* p = (const float4*)(src + (size_t)wid * D_DIM);
  float s = 0.f;
  #pragma unroll
  for (int i = lane; i < D_DIM / 4; i += 32) {
    float4 v = p[i];
    s += v.x * v.x + v.y * v.y + v.z * v.z + v.w * v.w;
  }
  #pragma unroll
  for (int off = 16; off > 0; off >>= 1) s += __shfl_down(s, off, 32);
  if (lane == 0) dst[wid] = s;
}

__global__ void tidal_temp(const float* __restrict__ temp,
                           const float* __restrict__ fscale,
                           const unsigned char* __restrict__ frozen,
                           float* __restrict__ rtemp, int n) {
  int i = blockIdx.x * blockDim.x + threadIdx.x;
  if (i < n) {
    float es = frozen[i] ? fscale[i] : MIN_SCALE;
    rtemp[i] = 1.0f / ((fabsf(temp[i]) + 0.1f) * es);
  }
}

__global__ void tidal_cvt(const float* __restrict__ s, __bf16* __restrict__ d,
                          int n4) {
  int i = blockIdx.x * blockDim.x + threadIdx.x;
  if (i >= n4) return;
  float4 v = ((const float4*)s)[i];
  union { __bf16 b[4]; uint2 u; } t;
  t.b[0] = (__bf16)v.x; t.b[1] = (__bf16)v.y;
  t.b[2] = (__bf16)v.z; t.b[3] = (__bf16)v.w;
  ((uint2*)d)[i] = t.u;
}

__global__ void tidal_cvt_t(const float* __restrict__ s, __bf16* __restrict__ d,
                            int rows, int cols) {
  int idx = blockIdx.x * blockDim.x + threadIdx.x;
  if (idx >= rows * cols) return;
  int r = idx / cols, c = idx - r * cols;
  d[(size_t)c * rows + r] = (__bf16)s[idx];  // d is [cols][rows]
}

// ---- Fused main kernel ----------------------------------------------------
// LDS map (98 KB):
//   [0     ,32768)  XP buf0: Xs0[128][64], Ps0[128][64]  (bf16)
//   [0     ,32768)  Ws [128][128] bf16 (weights)  <- aliases XP buf0
//   [32768 ,65536)  XP buf1: Xs1, Ps1
//   [65536 ,81920)  Vt buf0 [64][128] bf16
//   [81920 ,98304)  Vt buf1
//   [98304 ..     ] nxs[128], nps[128], rts[128], rsum[128] f32
__global__ __launch_bounds__(256, 1)
void tidal_main(const __bf16* __restrict__ xbf, const __bf16* __restrict__ pbf,
                const __bf16* __restrict__ vbfT, const float* __restrict__ nx,
                const float* __restrict__ npn, const float* __restrict__ rtp,
                float* __restrict__ out, int nTiles, int Mtot, int Ntot) {
  __shared__ __align__(16) unsigned char smem[100352];
  __bf16* Wsh  = (__bf16*)(smem);
  float*  nxs  = (float*)(smem + 98304);
  float*  nps  = (float*)(smem + 98816);
  float*  rts  = (float*)(smem + 99328);
  float*  rsum = (float*)(smem + 99840);
  const unsigned ldsBase = (unsigned)(size_t)(void*)smem;

  const int tid  = threadIdx.x;
  const int lane = tid & 31;
  const int wave = tid >> 5;
  const int l15  = lane & 15;
  const int h    = lane >> 4;
  const int m0   = blockIdx.x * MB;

  const f32x8 zero8 = {0.f, 0.f, 0.f, 0.f, 0.f, 0.f, 0.f, 0.f};
  f32x8 acc[32];
  #pragma unroll
  for (int i = 0; i < 32; ++i) acc[i] = zero8;
  float prsum[8] = {0.f, 0.f, 0.f, 0.f, 0.f, 0.f, 0.f, 0.f};

  if (tid < 128) { nxs[tid] = nx[m0 + tid]; rsum[tid] = 0.f; }

  for (int nt0 = 0; nt0 < nTiles; ++nt0) {
    const int nb0 = nt0 * NB;
    if (tid < 128) { nps[tid] = npn[nb0 + tid]; rts[tid] = rtp[nb0 + tid]; }

    // --- GEMM1 TDM prologue: chunks 0,1 -> buffers 0,1 (wave 0 only) ---
    if (wave == 0) {
      #pragma unroll
      for (int c = 0; c < 2; ++c) {
        tdm_load_2d_bf16(xbf + (size_t)m0 * D_DIM + c * KB,
                         ldsBase + c * 32768, D_DIM, Mtot, KB, MB);
        tdm_load_2d_bf16(pbf + (size_t)nb0 * D_DIM + c * KB,
                         ldsBase + c * 32768 + 16384, D_DIM, Ntot, KB, NB);
      }
    }

    f32x8 S[8];
    #pragma unroll
    for (int i = 0; i < 8; ++i) S[i] = zero8;

    auto gemm1_step = [&](const __bf16* Xb, const __bf16* Pb) {
      const __bf16* xrow = Xb + (wave * 16 + l15) * KB;
      #pragma unroll
      for (int s = 0; s < 2; ++s) {
        bf16x16 a = lds_fragA(xrow, s * 32, h);
        #pragma unroll
        for (int nt = 0; nt < 8; ++nt) {
          bf16x16 b = lds_fragB(Pb + (nt * 16 + l15) * KB, s * 32, h);
          S[nt] = __builtin_amdgcn_wmma_f32_16x16x32_bf16(
              false, a, false, b, (short)0, S[nt], false, false);
        }
      }
    };

    // --- GEMM1: double-buffered DMA / compute pipeline ---
    for (int kc = 0; kc < 7; ++kc) {
      if (wave == 0) __builtin_amdgcn_s_wait_tensorcnt(2);
      __syncthreads();
      const __bf16* Xb = (const __bf16*)(smem + (kc & 1) * 32768);
      gemm1_step(Xb, Xb + 16384);
      __syncthreads();
      if (wave == 0 && kc < 6) {
        tdm_load_2d_bf16(xbf + (size_t)m0 * D_DIM + (kc + 2) * KB,
                         ldsBase + (kc & 1) * 32768, D_DIM, Mtot, KB, MB);
        tdm_load_2d_bf16(pbf + (size_t)nb0 * D_DIM + (kc + 2) * KB,
                         ldsBase + (kc & 1) * 32768 + 16384, D_DIM, Ntot, KB, NB);
      }
    }
    if (wave == 0) __builtin_amdgcn_s_wait_tensorcnt(0);
    __syncthreads();
    {
      const __bf16* Xb = (const __bf16*)(smem + 32768);  // kc=7 -> buf1
      gemm1_step(Xb, Xb + 16384);
    }
    __syncthreads();  // all XP reads done; Ws (alias of buf0) may be written

    // --- GEMM2 TDM prologue (overlaps weight math): Vt chunks 0,1 ---
    if (wave == 0) {
      #pragma unroll
      for (int c = 0; c < 2; ++c)
        tdm_load_2d_bf16(vbfT + (size_t)c * DB * Ntot + nb0,
                         ldsBase + 65536 + c * 16384, Ntot, D_DIM, NB, DB);
    }

    // --- weights: w = exp(-sqrt(max(nx+np-2*dot,0)) / temp) ---
    float nxr[8];
    #pragma unroll
    for (int r = 0; r < 8; ++r) nxr[r] = nxs[wave * 16 + h * 8 + r];
    #pragma unroll
    for (int nt = 0; nt < 8; ++nt) {
      float npv = nps[nt * 16 + l15];
      float rtv = rts[nt * 16 + l15];
      #pragma unroll
      for (int r = 0; r < 8; ++r) {
        float d2 = nxr[r] + npv - 2.0f * S[nt][r];
        float w  = __expf(-__fsqrt_rn(fmaxf(d2, 0.0f)) * rtv);
        prsum[r] += w;
        Wsh[(wave * 16 + h * 8 + r) * NB + nt * 16 + l15] = (__bf16)w;
      }
    }
    __syncthreads();

    bf16x16 a2[4];
    const __bf16* wrow = Wsh + (wave * 16 + l15) * NB;
    #pragma unroll
    for (int ks = 0; ks < 4; ++ks) a2[ks] = lds_fragA(wrow, ks * 32, h);

    // --- GEMM2: acc += W . V, double-buffered Vt DMA ---
    #pragma unroll
    for (int dc = 0; dc < 8; ++dc) {
      if (wave == 0) {
        if (dc == 7) __builtin_amdgcn_s_wait_tensorcnt(0);
        else         __builtin_amdgcn_s_wait_tensorcnt(1);
      }
      __syncthreads();
      const __bf16* Vb = (const __bf16*)(smem + 65536 + (dc & 1) * 16384);
      #pragma unroll
      for (int dt = 0; dt < 4; ++dt) {
        #pragma unroll
        for (int ks = 0; ks < 4; ++ks) {
          bf16x16 b = lds_fragB(Vb + (dt * 16 + l15) * NB, ks * 32, h);
          acc[dc * 4 + dt] = __builtin_amdgcn_wmma_f32_16x16x32_bf16(
              false, a2[ks], false, b, (short)0, acc[dc * 4 + dt], false, false);
        }
      }
      __syncthreads();
      if (wave == 0 && dc < 6)
        tdm_load_2d_bf16(vbfT + (size_t)(dc + 2) * DB * Ntot + nb0,
                         ldsBase + 65536 + (dc & 1) * 16384, Ntot, D_DIM, NB, DB);
    }
  }

  // ---- row-sum reduction + normalize + store ----
  #pragma unroll
  for (int r = 0; r < 8; ++r)
    atomicAdd(&rsum[wave * 16 + h * 8 + r], prsum[r]);
  __syncthreads();
  #pragma unroll
  for (int r = 0; r < 8; ++r) {
    float inv = 1.0f / (rsum[wave * 16 + h * 8 + r] + 1e-8f);
    size_t row = (size_t)(m0 + wave * 16 + h * 8 + r) * D_DIM;
    #pragma unroll
    for (int ct = 0; ct < 32; ++ct)
      out[row + ct * 16 + l15] = acc[ct][r] * inv;
  }
}

extern "C" void kernel_launch(void* const* d_in, const int* in_sizes, int n_in,
                              void* d_out, int out_size, void* d_ws, size_t ws_size,
                              hipStream_t stream) {
  const float* x    = (const float*)d_in[0];
  const float* pos  = (const float*)d_in[1];
  const float* val  = (const float*)d_in[2];
  const float* temp = (const float*)d_in[3];
  const float* fsc  = (const float*)d_in[4];
  const unsigned char* frz = (const unsigned char*)d_in[5];

  const int M = in_sizes[0] / D_DIM;   // 16384
  const int N = in_sizes[3];           // 4096

  // workspace: nx[M] np[N] rt[N] | xbf[M*512] pbf[N*512] vbfT[512*N] (bf16)
  float* nx = (float*)d_ws;
  float* np = nx + M;
  float* rt = np + N;
  __bf16* xbf  = (__bf16*)((char*)d_ws + 131072);
  __bf16* pbf  = xbf + (size_t)M * D_DIM;
  __bf16* vbfT = pbf + (size_t)N * D_DIM;

  tidal_rownorm<<<(M * 32 + 255) / 256, 256, 0, stream>>>(x, nx, M);
  tidal_rownorm<<<(N * 32 + 255) / 256, 256, 0, stream>>>(pos, np, N);
  tidal_temp<<<(N + 255) / 256, 256, 0, stream>>>(temp, fsc, frz, rt, N);
  tidal_cvt<<<((M * D_DIM / 4) + 255) / 256, 256, 0, stream>>>(x, xbf, M * D_DIM / 4);
  tidal_cvt<<<((N * D_DIM / 4) + 255) / 256, 256, 0, stream>>>(pos, pbf, N * D_DIM / 4);
  tidal_cvt_t<<<((N * D_DIM) + 255) / 256, 256, 0, stream>>>(val, vbfT, N, D_DIM);

  tidal_main<<<M / MB, 256, 0, stream>>>(xbf, pbf, vbfT, nx, np, rt,
                                         (float*)d_out, N / NB, M, N);
}
